// MultiHeadAttention_5317169512877
// MI455X (gfx1250) — matrix operand order repaired and checked by reading, not compiled, not run on hardware
//
#include <hip/hip_runtime.h>
#include <hip/hip_bf16.h>
#include <stdint.h>

typedef __bf16 bf16;
typedef __attribute__((ext_vector_type(16))) __bf16 v16bf;
typedef __attribute__((ext_vector_type(8)))  float  v8f;

#define BB  8
#define SS  1024
#define DD  768
#define HH  12
#define DKK 64

// LDS row strides (padded to stagger banks for WMMA b128 fragment reads)
#define SCS 1026   // f32 score row stride
#define PSD 1032   // bf16 prob row stride

union FragB16 { v16bf v; uint4 q[2]; };
union Pack8   { bf16 h[8]; uint4 q; };

// A-fragment (16x32 bf16): lane = {half = l>>4, m = l&15}; 16 contiguous
// K: K = k0 + half*16 .. +16
static __device__ __forceinline__ v16bf frag_a_bf16(const bf16* row, int k0, int half) {
  FragB16 f;
  f.q[0] = *(const uint4*)(row + k0 + half * 16);
  f.q[1] = *(const uint4*)(row + k0 + half * 16 + 8);
  return f.v;
}
// B-fragment (32x16 bf16): lane = column n=l&15; 16 contiguous K at k0 + (l>>4)*16
static __device__ __forceinline__ v16bf frag_b_bf16(const bf16* p) {
  FragB16 f;
  f.q[0] = ((const uint4*)p)[0];
  f.q[1] = ((const uint4*)p)[1];
  return f.v;
}
// Same fragments built from f32 memory with on-the-fly bf16 conversion
static __device__ __forceinline__ v16bf frag_a_f32(const float* row, int k0, int half) {
  v16bf a;
#pragma unroll
  for (int r = 0; r < 2; ++r) {
    const float* p = row + k0 + half * 16 + r * 8;
    float4 f0 = *(const float4*)p;
    float4 f1 = *(const float4*)(p + 4);
    a[r*8+0] = (bf16)f0.x; a[r*8+1] = (bf16)f0.y; a[r*8+2] = (bf16)f0.z; a[r*8+3] = (bf16)f0.w;
    a[r*8+4] = (bf16)f1.x; a[r*8+5] = (bf16)f1.y; a[r*8+6] = (bf16)f1.z; a[r*8+7] = (bf16)f1.w;
  }
  return a;
}
static __device__ __forceinline__ v16bf frag_b_f32(const float* p) {
  v16bf b;
#pragma unroll
  for (int i = 0; i < 4; ++i) {
    float4 f = ((const float4*)p)[i];
    b[i*4+0] = (bf16)f.x; b[i*4+1] = (bf16)f.y; b[i*4+2] = (bf16)f.z; b[i*4+3] = (bf16)f.w;
  }
  return b;
}

// ---------------------------------------------------------------------------
// Kernel 1: QKV projections.  y = x @ W^T  (y[i,j] = dot(x_i, W_j); W row-major
// so B-fragment columns are contiguous in memory).  Q,K stored bf16 row-major
// [B*S, D]; V stored bf16 TRANSPOSED as [B,H,DK,S] via an LDS bounce so the
// PV-GEMM B-fragment is a contiguous 32B load per lane.
// Tile: 128 tokens x 64 dims (= exactly one head) per 256-thread block.
// ---------------------------------------------------------------------------
__global__ __launch_bounds__(256) void qkv_kernel(
    const float* __restrict__ x,
    const float* __restrict__ Wq, const float* __restrict__ Wk, const float* __restrict__ Wv,
    bf16* __restrict__ yq, bf16* __restrict__ yk, bf16* __restrict__ vt)
{
  __shared__ bf16 vtile[128][72];  // 64 cols padded to 72 for the transpose bounce

  const int z      = blockIdx.z;                 // 0=Q 1=K 2=V
  const float* __restrict__ W = (z == 0) ? Wq : ((z == 1) ? Wk : Wv);
  const int m_base = blockIdx.x * 128;           // token rows
  const int h      = blockIdx.y;                 // head == 64-wide N block
  const int n_base = h * DKK;
  const int tid  = threadIdx.x;
  const int wave = tid >> 5, lane = tid & 31, half = lane >> 4, l16 = lane & 15;
  const int m0   = m_base + wave * 16;

  v8f acc[4] = {};
  const float* arow = x + (size_t)(m0 + l16) * DD;

  for (int k0 = 0; k0 < DD; k0 += 32) {
    __builtin_prefetch(arow + k0 + 64);          // global_prefetch_b8
    v16bf a = frag_a_f32(arow, k0, half);
#pragma unroll
    for (int nt = 0; nt < 4; ++nt) {
      const float* bp = W + (size_t)(n_base + nt * 16 + l16) * DD + k0 + half * 16;
      v16bf b = frag_b_f32(bp);
      acc[nt] = __builtin_amdgcn_wmma_f32_16x16x32_bf16(
          false, a, false, b, (short)0, acc[nt], false, false);
    }
  }

  if (z < 2) {
    bf16* __restrict__ y = (z == 0) ? yq : yk;
#pragma unroll
    for (int nt = 0; nt < 4; ++nt)
#pragma unroll
      for (int g = 0; g < 8; ++g) {
        int row = m0 + half * 8 + g;             // C layout: M = g + half*8
        int col = n_base + nt * 16 + l16;        // N = lane&15
        y[(size_t)row * DD + col] = (bf16)acc[nt][g];
      }
  } else {
    // stage tile, then write V transposed (coalesced along S)
#pragma unroll
    for (int nt = 0; nt < 4; ++nt)
#pragma unroll
      for (int g = 0; g < 8; ++g)
        vtile[wave * 16 + half * 8 + g][nt * 16 + l16] = (bf16)acc[nt][g];
    __syncthreads();
    const int b     = m_base >> 10;              // 1024 tokens per batch
    const int sbase = m_base & (SS - 1);
    const int dk    = tid >> 2;                  // 0..63
    const int seg   = tid & 3;                   // 32-token segment
    bf16* __restrict__ dst =
        vt + (((size_t)b * HH + h) * DKK + dk) * SS + sbase + seg * 32;
#pragma unroll
    for (int j = 0; j < 4; ++j) {
      Pack8 pk;
#pragma unroll
      for (int e = 0; e < 8; ++e) pk.h[e] = vtile[seg * 32 + j * 8 + e][dk];
      *(uint4*)(dst + j * 8) = pk.q;
    }
  }
}

// ---------------------------------------------------------------------------
// Kernel 2: attention for one (b, h, 32-query tile).  Full 32x1024 score tile
// lives in LDS (CDNA5's 320KB WGP LDS makes this a single-pass softmax).
//  GEMM1: scores = Q K^T / 8   (8 waves: 2 q-halves x 4 key-blocks of 256)
//  softmax row-wise, probs -> d_out (non-temporal, 402MB stream) + bf16 LDS
//  GEMM2: out = P V            (8 waves: one 16x16 tile each)
// ---------------------------------------------------------------------------
__global__ __launch_bounds__(256) void attn_kernel(
    const bf16* __restrict__ yq, const bf16* __restrict__ yk,
    const bf16* __restrict__ vt, const unsigned char* __restrict__ mask,
    float* __restrict__ scores_out, float* __restrict__ attn)
{
  extern __shared__ char smem[];
  float* sc = (float*)smem;                              // [32][SCS] f32
  bf16*  pp = (bf16*)(smem + 32 * SCS * sizeof(float));  // [32][PSD] bf16

  const int q0 = blockIdx.x * 32;
  const int h  = blockIdx.y;
  const int b  = blockIdx.z;
  const int tid  = threadIdx.x;
  const int wave = tid >> 5, lane = tid & 31, half = lane >> 4, l16 = lane & 15;
  const int mh = wave >> 2;   // which 16-row q half
  const int kb = wave & 3;    // which 256-key block

  // ---- GEMM1: Q(32x64) x K^T -> scores(32x1024) ----
  {
    v8f acc[16] = {};
    const bf16* qrow = yq + (size_t)(b * SS + q0 + mh * 16 + l16) * DD + h * DKK;
#pragma unroll
    for (int k0 = 0; k0 < DKK; k0 += 32) {
      v16bf a = frag_a_bf16(qrow, k0, half);
#pragma unroll
      for (int nt = 0; nt < 16; ++nt) {
        const bf16* kp = yk + (size_t)(b * SS + kb * 256 + nt * 16 + l16) * DD
                         + h * DKK + k0 + half * 16;
        v16bf bf = frag_b_bf16(kp);
        acc[nt] = __builtin_amdgcn_wmma_f32_16x16x32_bf16(
            false, a, false, bf, (short)0, acc[nt], false, false);
      }
    }
    const float scale = 0.125f;  // 1/sqrt(64)
#pragma unroll
    for (int nt = 0; nt < 16; ++nt)
#pragma unroll
      for (int g = 0; g < 8; ++g) {
        int m   = mh * 16 + half * 8 + g;
        int key = kb * 256 + nt * 16 + l16;
        float v = acc[nt][g] * scale;
        if (!mask[b * SS + key]) v = -__builtin_inff();  // key-side mask
        sc[m * SCS + key] = v;
      }
  }
  __syncthreads();

  // ---- softmax: each wave owns 4 rows; 32 lanes sweep 1024 cols ----
  for (int rr = 0; rr < 4; ++rr) {
    const int r = wave * 4 + rr;
    float* row = sc + r * SCS;
    float mx = -__builtin_inff();
#pragma unroll
    for (int j = 0; j < 32; ++j) mx = fmaxf(mx, row[j * 32 + lane]);
#pragma unroll
    for (int o = 16; o > 0; o >>= 1) mx = fmaxf(mx, __shfl_xor(mx, o, 32));
    float eb[32];
    float ssum = 0.f;
#pragma unroll
    for (int j = 0; j < 32; ++j) {
      float e = __expf(row[j * 32 + lane] - mx);
      eb[j] = e;
      ssum += e;
    }
#pragma unroll
    for (int o = 16; o > 0; o >>= 1) ssum += __shfl_xor(ssum, o, 32);
    const float inv = 1.0f / ssum;
    float* so = scores_out + (((size_t)(b * HH + h) * SS) + q0 + r) * SS;
#pragma unroll
    for (int j = 0; j < 32; ++j) {
      float p = eb[j] * inv;
      pp[r * PSD + j * 32 + lane] = (bf16)p;
      __builtin_nontemporal_store(p, so + j * 32 + lane);  // 402MB stream: bypass caches
    }
  }
  __syncthreads();

  // ---- GEMM2: P(32x1024) x V(1024x64), one 16x16 tile per wave ----
  {
    v8f acc = {};
    const int nb = wave & 3;  // dk block of 16
    const bf16* prow = pp + (size_t)(mh * 16 + l16) * PSD;
    const bf16* vcol = vt + (((size_t)b * HH + h) * DKK + nb * 16 + l16) * SS;
#pragma unroll 4
    for (int k0 = 0; k0 < SS; k0 += 32) {
      v16bf a  = frag_a_bf16(prow, k0, half);          // from LDS (b128 reads)
      v16bf bf = frag_b_bf16(vcol + k0 + half * 16);   // V^T: contiguous keys
      acc = __builtin_amdgcn_wmma_f32_16x16x32_bf16(
          false, a, false, bf, (short)0, acc, false, false);
    }
#pragma unroll
    for (int g = 0; g < 8; ++g) {
      int s = q0 + mh * 16 + half * 8 + g;
      int d = h * DKK + nb * 16 + l16;
      attn[((size_t)b * SS + s) * DD + d] = acc[g];
    }
  }
}

// ---------------------------------------------------------------------------
// Kernel 3: residual + LayerNorm, one 256-thread block per token row (3 elems
// per thread), wave32 shuffle + LDS cross-wave reduction.
// ---------------------------------------------------------------------------
__global__ __launch_bounds__(256) void ln_kernel(
    const float* __restrict__ attn, const float* __restrict__ x,
    const float* __restrict__ gamma, const float* __restrict__ beta,
    float* __restrict__ out)
{
  __shared__ float sred[8];
  __shared__ float sbc;
  const int row = blockIdx.x;
  const int tid = threadIdx.x, wave = tid >> 5, lane = tid & 31;
  const float* ar = attn + (size_t)row * DD;
  const float* xr = x + (size_t)row * DD;

  float v[3];
#pragma unroll
  for (int j = 0; j < 3; ++j) {
    int c = tid + j * 256;
    v[j] = ar[c] + xr[c];
  }
  float p = v[0] + v[1] + v[2];
#pragma unroll
  for (int o = 16; o > 0; o >>= 1) p += __shfl_xor(p, o, 32);
  if (lane == 0) sred[wave] = p;
  __syncthreads();
  if (tid == 0) { float t = 0; for (int i = 0; i < 8; ++i) t += sred[i]; sbc = t; }
  __syncthreads();
  const float mu = sbc * (1.0f / DD);

  float q = 0.f;
#pragma unroll
  for (int j = 0; j < 3; ++j) { float d = v[j] - mu; q += d * d; }
#pragma unroll
  for (int o = 16; o > 0; o >>= 1) q += __shfl_xor(q, o, 32);
  if (lane == 0) sred[wave] = q;
  __syncthreads();
  if (tid == 0) { float t = 0; for (int i = 0; i < 8; ++i) t += sred[i]; sbc = t; }
  __syncthreads();
  const float rstd = rsqrtf(sbc * (1.0f / DD) + 1e-6f);

#pragma unroll
  for (int j = 0; j < 3; ++j) {
    int c = tid + j * 256;
    float o = (v[j] - mu) * rstd * gamma[c] + beta[c];
    __builtin_nontemporal_store(o, out + (size_t)row * DD + c);
  }
}

// ---------------------------------------------------------------------------
extern "C" void kernel_launch(void* const* d_in, const int* in_sizes, int n_in,
                              void* d_out, int out_size, void* d_ws, size_t ws_size,
                              hipStream_t stream)
{
  const float* x            = (const float*)d_in[0];
  const unsigned char* mask = (const unsigned char*)d_in[1];
  const float* Wq           = (const float*)d_in[2];
  const float* Wk           = (const float*)d_in[3];
  const float* Wv           = (const float*)d_in[4];
  const float* gamma        = (const float*)d_in[5];
  const float* beta         = (const float*)d_in[6];

  // workspace: Q,K bf16 [B*S,D]; V^T bf16 [B,H,DK,S]; attn f32 [B,S,D]  (~63MB)
  bf16* yq   = (bf16*)d_ws;
  bf16* yk   = yq + (size_t)BB * SS * DD;
  bf16* vt   = yk + (size_t)BB * SS * DD;
  float* attn = (float*)(vt + (size_t)BB * SS * DD);

  float* scores_out = (float*)d_out;                       // [B,H,S,S] f32
  float* out2       = scores_out + (size_t)BB * HH * SS * SS;  // [B,S,D] f32

  qkv_kernel<<<dim3(BB * SS / 128, HH, 3), 256, 0, stream>>>(x, Wq, Wk, Wv, yq, yk, vt);

  const size_t smem = 32 * SCS * sizeof(float) + 32 * PSD * sizeof(bf16);  // ~193KB of 320KB LDS
  attn_kernel<<<dim3(SS / 32, HH, BB), 256, smem, stream>>>(yq, yk, vt, mask, scores_out, attn);

  ln_kernel<<<BB * SS, 256, 0, stream>>>(attn, x, gamma, beta, out2);
}
